// ResNetBasicblock_60859686584482
// MI455X (gfx1250) — matrix operand order repaired and checked by reading, not compiled, not run on hardware
//
#include <hip/hip_runtime.h>

typedef __attribute__((ext_vector_type(16))) _Float16 v16h;
typedef __attribute__((ext_vector_type(8)))  _Float16 v8h;
typedef __attribute__((ext_vector_type(8)))  float    v8f;
typedef __attribute__((ext_vector_type(4)))  int      v4i;

#define N_IMG 32
#define C 64
#define H 128
#define W 128
#define K_W_MULT 2.0f
#define GAMMA_PR 0.3f
#define BN_EPS   1e-5f
#define HW_CNT   ((float)(N_IMG * H * W))   // per-channel BN count

// ---- workspace layout (bytes) ----
#define WFRAG_BYTES (9 * 2 * 4 * 32 * 16 * 2)        // 72 fragments * 32 lanes * 16 f16 = 73728
#define OFF_WFRAG_A 0
#define OFF_WFRAG_B (OFF_WFRAG_A + WFRAG_BYTES)
#define OFF_STATS   (OFF_WFRAG_B + WFRAG_BYTES)      // 256 floats: sum1,sq1,sum2,sq2
#define OFF_SCALEB  (OFF_STATS + 256 * 4)            // 256 floats: scale1,bias1,scale2,bias2
#define OFF_RAW1    (OFF_SCALEB + 256 * 4)           // 149504, 256B aligned
#define RAW_BYTES   ((size_t)N_IMG * H * W * C * 2)  // NHWC f16: 67,108,864
#define OFF_RAW2    (OFF_RAW1 + RAW_BYTES)

// ---- CDNA5 async global->LDS path (guarded; falls back to manual staging) ----
#define AS1 __attribute__((address_space(1)))
#define AS3 __attribute__((address_space(3)))

#if defined(__has_builtin)
# if __has_builtin(__builtin_amdgcn_global_load_async_to_lds_b128) && \
     __has_builtin(__builtin_amdgcn_s_wait_asynccnt)
#  define HAVE_ASYNC 1
# endif
#endif
#ifndef HAVE_ASYNC
# define HAVE_ASYNC 0
#endif

#if HAVE_ASYNC
__device__ __forceinline__ AS1 char* as1_ptr(const void* p) {
    // generic global pointer value == addrspace(1) value
    return (AS1 char*)(unsigned long long)p;
}
__device__ __forceinline__ AS3 char* as3_ptr(void* p) {
    // low 32 bits of a generic LDS pointer are the LDS offset (flat aperture rule)
    return (AS3 char*)(unsigned int)(unsigned long long)p;
}
// builtin signature: (v4i addrspace(1)*, v4i addrspace(3)*, imm offset, imm cpol)
__device__ __forceinline__ void async_b128(AS1 char* g, AS3 char* l) {
    __builtin_amdgcn_global_load_async_to_lds_b128((AS1 v4i*)g, (AS3 v4i*)l, 0, 0);
}
#endif

// ---------------------------------------------------------------------------
// Zero BN stat accumulators (graph-capture-safe replacement for memset)
// ---------------------------------------------------------------------------
__global__ void zero_stats_kernel(float* __restrict__ stats) {
    stats[threadIdx.x] = 0.0f;
}

// ---------------------------------------------------------------------------
// Weight prep: structured channel-group prune + 2-bit quant ({-1,-1/3,1/3,1}*a)
// and re-layout into WMMA B-fragment-ready form:
//   frag f = tap*8 + kc*4 + nt ; within fragment: [lane 0..31][e 0..15] f16
//   lane<16 : N=lane,    K={0..7,16..23}[e] ; lane>=16 : N=lane-16, K={8..15,24..31}[e]
//   cin = kc*32 + K ; cout = nt*16 + N ; (dh,dw) = (tap/3, tap%3)
// ---------------------------------------------------------------------------
__global__ __launch_bounds__(256) void prep_weights_kernel(
    const float* __restrict__ w_a, const float* __restrict__ w_b,
    _Float16* __restrict__ fragA, _Float16* __restrict__ fragB)
{
    const float*  w    = (blockIdx.x == 0) ? w_a   : w_b;
    _Float16*     frag = (blockIdx.x == 0) ? fragA : fragB;
    const int t = threadIdx.x;

    __shared__ float norms[256];
    __shared__ float red[256];
    __shared__ float s_thr, s_alpha, s_inv_alpha;

    // one thread = one (cout, channel-group) of 16 cin * 9 taps
    {
        const int o = t >> 2, gi = t & 3;
        float ss = 0.f;
        for (int ci = gi * 16; ci < gi * 16 + 16; ++ci)
            for (int k = 0; k < 9; ++k) {
                float v = w[(o * 64 + ci) * 9 + k];
                ss += v * v;
            }
        norms[t] = sqrtf(ss);
        red[t]   = norms[t];
    }
    __syncthreads();
    for (int s = 128; s > 0; s >>= 1) {
        if (t < s) red[t] += red[t + s];
        __syncthreads();
    }
    if (t == 0) s_thr = GAMMA_PR * (red[0] * (1.0f / 256.0f));
    __syncthreads();

    // alpha_w = 2 * mean|w_pruned| (zeros of pruned groups stay in denominator)
    {
        const int o = t >> 2, gi = t & 3;
        float asum = 0.f;
        if (norms[t] >= s_thr)
            for (int ci = gi * 16; ci < gi * 16 + 16; ++ci)
                for (int k = 0; k < 9; ++k)
                    asum += fabsf(w[(o * 64 + ci) * 9 + k]);
        red[t] = asum;
    }
    __syncthreads();
    for (int s = 128; s > 0; s >>= 1) {
        if (t < s) red[t] += red[t + s];
        __syncthreads();
    }
    if (t == 0) {
        float alpha = K_W_MULT * (red[0] * (1.0f / 36864.0f));
        s_alpha     = alpha;
        s_inv_alpha = (alpha > 0.f) ? 1.f / alpha : 0.f;
    }
    __syncthreads();

    for (int e = t; e < 36864; e += 256) {
        const int f    = e >> 9;
        const int r    = e & 511;
        const int lane = r >> 4, el = r & 15;
        const int tap  = f >> 3;
        const int kc   = (f >> 2) & 1;
        const int nt   = f & 3;
        const int K    = ((lane >> 4) * 8) + ((el < 8) ? el : (8 + el));
        const int ci   = kc * 32 + K;
        const int o    = nt * 16 + (lane & 15);
        float wv = w[(o * 64 + ci) * 9 + tap];
        if (norms[o * 4 + (ci >> 4)] < s_thr) wv = 0.f;
        float q = wv * s_inv_alpha;
        q = fminf(fmaxf(q, -1.f), 1.f);
        q = rintf((q + 1.f) * 1.5f) * (1.f / 1.5f) - 1.f;     // {-1,-1/3,1/3,1}
        frag[e] = (_Float16)(s_alpha * q);
    }
}

// ---------------------------------------------------------------------------
// Implicit-GEMM 3x3 conv via v_wmma_f32_16x16x32_f16.
// Block = (n, h): 128 output w positions x 64 cout. 8 wave32s, each wave owns
// a 16-row M tile and all four 16-col N tiles. K = 64cin * 9taps, consumed as
// 9 taps x 2 chunks of 32, FULLY UNROLLED (72 wmma) so accumulators stay
// pinned in place (no cross-tuple copies -> no WMMA->VALU hazard NOPs).
//  - B fragments (all 72, 72KB) staged to LDS once per block
//  - Weights (and, for the f16 NHWC conv2 input, the halo tile) staged with
//    GLOBAL_LOAD_ASYNC_TO_LDS_B128 when available (ASYNCcnt tracked)
//  - Per-channel BN sum/sumsq folded via ds_add_f32 -> 1 global atomic/ch/block
// ---------------------------------------------------------------------------
template<bool IN_F32_NCHW>
__global__ __launch_bounds__(256) void conv3x3_wmma_kernel(
    const void* __restrict__ in_raw,
    const _Float16* __restrict__ wfrag,
    _Float16* __restrict__ out_raw,          // NHWC f16
    float* __restrict__ gSum, float* __restrict__ gSq)
{
    __shared__ __align__(16) _Float16 Alds[3 * 130 * 64];   // 49,920 B halo tile
    __shared__ __align__(32) _Float16 Blds[36864];          // 73,728 B weight frags
    __shared__ float sSum[64], sSq[64];

    const int t    = threadIdx.x;
    const int lane = t & 31;
    const int wv   = t >> 5;           // wave id 0..7 (wave32)
    const int h    = blockIdx.x;
    const int n    = blockIdx.y;

    if (t < 64) { sSum[t] = 0.f; sSq[t] = 0.f; }

    const float*    inF = (const float*)in_raw;
    const _Float16* inH = (const _Float16*)in_raw;

    // ---- stage weight fragments to LDS (async when available) ----
#if HAVE_ASYNC
    {
        AS1 char* g = as1_ptr(wfrag);
        AS3 char* l = as3_ptr(Blds);
        for (int off = t * 16; off < WFRAG_BYTES; off += 256 * 16)
            async_b128(g + off, l + off);
    }
#else
    for (int e = t * 8; e < 36864; e += 256 * 8)
        *(v8h*)&Blds[e] = *(const v8h*)&wfrag[e];
#endif

    // ---- stage input halo tile [row3][pos130][cin64] into LDS ----
    if (IN_F32_NCHW) {
        // f32 NCHW input: must convert, manual path (overlaps async weight DMA)
        if (t == 0 && h + 1 < H)
            __builtin_prefetch(&inF[((size_t)(n * C) * H + (h + 1)) * W], 0, 1);
        for (int e = t; e < 3 * 130 * 64; e += 256) {
            int c = e / 390; int rp = e - c * 390;
            int r = rp / 130, p = rp - r * 130;     // consecutive t -> consecutive w
            const int hg = h + r - 1, wg = p - 1;
            _Float16 val = (_Float16)0.f;
            if (hg >= 0 && hg < H && wg >= 0 && wg < W)
                val = (_Float16)inF[((size_t)(n * C + c) * H + hg) * W + wg];
            Alds[(r * 130 + p) * 64 + c] = val;
        }
    } else {
#if HAVE_ASYNC
        // f16 NHWC input: zero halo precisely, async-copy the interior rows
        v8h z = {};
        if (t < 48) {   // columns p=0 and p=129 of all 3 rows
            const int r = t / 16, side = (t >> 3) & 1, c8 = (t & 7) * 8;
            const int p = side ? 129 : 0;
            *(v8h*)&Alds[(r * 130 + p) * 64 + c8] = z;
        }
        AS1 char* gin = as1_ptr(inH);
        AS3 char* lA  = as3_ptr(Alds);
        for (int r = 0; r < 3; ++r) {
            const int hg = h + r - 1;
            if (hg < 0 || hg >= H) {
                for (int e = t * 8; e < 130 * 64; e += 256 * 8)
                    *(v8h*)&Alds[r * 130 * 64 + e] = z;     // OOB row -> zeros
            } else {
                // 128 positions * 64ch * 2B = 16KB contiguous in global and LDS
                const size_t gbase = ((size_t)(n * H + hg) * W) * C * 2;
                const int    lbase = (r * 130 + 1) * 64 * 2;
                for (int off = t * 16; off < 128 * 64 * 2; off += 256 * 16)
                    async_b128(gin + gbase + off, lA + lbase + off);
            }
        }
#else
        for (int e = t; e < 3 * 130 * 64; e += 256) {
            int pos = e >> 6, c = e & 63;
            int r = pos / 130, p = pos - r * 130;   // consecutive t -> consecutive cin
            const int hg = h + r - 1, wg = p - 1;
            _Float16 val = (_Float16)0.f;
            if (hg >= 0 && hg < H && wg >= 0 && wg < W)
                val = inH[((size_t)(n * H + hg) * W + wg) * C + c];
            Alds[(r * 130 + p) * 64 + c] = val;
        }
#endif
    }

#if HAVE_ASYNC
    __builtin_amdgcn_s_wait_asynccnt(0);   // all async LDS writes landed
#endif
    __syncthreads();

    // ---- main WMMA loop: pure LDS feed, fully unrolled ----
    v8f acc[4] = {};
    const int lhalf  = lane & 15;     // M row (A/D) and N col (B/D) selector
    const int hihalf = lane >> 4;

    #pragma unroll
    for (int tap = 0; tap < 9; ++tap) {
        const int dh = tap / 3, dw = tap - dh * 3;
        #pragma unroll
        for (int kc = 0; kc < 2; ++kc) {
            // A fragment (16x32 f16): lane<16 holds K {0..7,16..23}, lane>=16 {8..15,24..31}
            const int pos = wv * 16 + lhalf + dw;
            const int c0  = kc * 32 + (hihalf ? 8 : 0);
            const _Float16* ap = &Alds[(dh * 130 + pos) * 64 + c0];
            v8h alo = *(const v8h*)ap;          // ds_load_b128
            v8h ahi = *(const v8h*)(ap + 16);   // ds_load_b128
            v16h a;
            #pragma unroll
            for (int i = 0; i < 8; ++i) { a[i] = alo[i]; a[8 + i] = ahi[i]; }

            #pragma unroll
            for (int nt = 0; nt < 4; ++nt) {
                const int f = tap * 8 + kc * 4 + nt;
                v16h b = *(const v16h*)&Blds[(f << 9) + (lane << 4)];  // 32B/lane LDS
                acc[nt] = __builtin_amdgcn_wmma_f32_16x16x32_f16(
                    false, a, false, b, (short)0, acc[nt], false, false);
            }
        }
    }

    // ---- epilogue: NHWC f16 raw output + per-channel BN partials ----
    #pragma unroll
    for (int nt = 0; nt < 4; ++nt) {
        const int cout = nt * 16 + lhalf;       // D: N = lane%16
        float s = 0.f, s2 = 0.f;
        #pragma unroll
        for (int j = 0; j < 8; ++j) {
            const int m  = j + hihalf * 8;      // D: VGPR j -> M=j (lanes<16) / 8+j
            const int wo = wv * 16 + m;
            const float v = acc[nt][j];
            out_raw[((size_t)(n * H + h) * W + wo) * C + cout] = (_Float16)v;
            s += v; s2 += v * v;
        }
        atomicAdd(&sSum[cout], s);              // ds_add_f32
        atomicAdd(&sSq[cout],  s2);
    }
    __syncthreads();
    if (t < 64) {
        atomicAdd(&gSum[t], sSum[t]);           // global_atomic_add_f32
        atomicAdd(&gSq[t],  sSq[t]);
    }
}

// ---------------------------------------------------------------------------
// Per-channel BN scale/bias from accumulated batch statistics
// ---------------------------------------------------------------------------
__global__ void bn_finalize_kernel(
    const float* __restrict__ sum, const float* __restrict__ sq,
    const float* __restrict__ gamma, const float* __restrict__ beta,
    float* __restrict__ scale, float* __restrict__ bias)
{
    const int t = threadIdx.x;
    if (t < 64) {
        const float inv  = 1.0f / HW_CNT;
        const float mean = sum[t] * inv;
        const float var  = sq[t] * inv - mean * mean;
        const float rs   = rsqrtf(var + BN_EPS);
        const float sc   = gamma[t] * rs;
        scale[t] = sc;
        bias[t]  = beta[t] - mean * sc;
    }
}

// ---------------------------------------------------------------------------
// BN + PACT 2-bit clipped-ReLU quant (forward value == quantized q), in place,
// v8h vectorized over NHWC f16
// ---------------------------------------------------------------------------
__global__ __launch_bounds__(256) void bn_act_kernel(
    _Float16* __restrict__ raw, const float* __restrict__ scale,
    const float* __restrict__ bias, const float* __restrict__ alpha_p)
{
    const size_t i0 = ((size_t)blockIdx.x * 256 + threadIdx.x) * 8;
    const float a   = alpha_p[0];
    const float qs  = 3.0f / a;            // (2^2 - 1) / alpha
    const float iqs = a * (1.0f / 3.0f);
    v8h v = *(v8h*)(raw + i0);
    #pragma unroll
    for (int i = 0; i < 8; ++i) {
        const int c = (int)((i0 + i) & 63);
        float f  = (float)v[i] * scale[c] + bias[c];
        float xc = fminf(fmaxf(f, 0.f), a);
        v[i] = (_Float16)(rintf(xc * qs) * iqs);
    }
    *(v8h*)(raw + i0) = v;
}

// ---------------------------------------------------------------------------
// Final: BN2 + act-quant on conv2 raw (NHWC f16) + residual x, NCHW fp32 out
// ---------------------------------------------------------------------------
__global__ __launch_bounds__(256) void residual_out_kernel(
    const float* __restrict__ x, const _Float16* __restrict__ raw2,
    const float* __restrict__ scale, const float* __restrict__ bias,
    const float* __restrict__ alpha_p, float* __restrict__ out)
{
    const size_t i0 = ((size_t)blockIdx.x * 256 + threadIdx.x) * 4;  // NCHW flat
    const int w0 = (int)(i0 % W);
    const int hq = (int)((i0 / W) % H);
    const int cq = (int)((i0 / ((size_t)W * H)) % C);
    const int nq = (int)(i0 / ((size_t)W * H * C));

    const float a   = alpha_p[0];
    const float qs  = 3.0f / a;
    const float iqs = a * (1.0f / 3.0f);
    const float sc  = scale[cq], bs = bias[cq];

    const float4 xi = *(const float4*)(x + i0);
    float r[4] = {xi.x, xi.y, xi.z, xi.w};
    #pragma unroll
    for (int i = 0; i < 4; ++i) {
        const size_t idx = ((size_t)(nq * H + hq) * W + (w0 + i)) * C + cq;
        float f  = (float)raw2[idx] * sc + bs;
        float xc = fminf(fmaxf(f, 0.f), a);
        r[i] += rintf(xc * qs) * iqs;
    }
    float4 o4 = {r[0], r[1], r[2], r[3]};
    *(float4*)(out + i0) = o4;
}

// ---------------------------------------------------------------------------
extern "C" void kernel_launch(void* const* d_in, const int* in_sizes, int n_in,
                              void* d_out, int out_size, void* d_ws, size_t ws_size,
                              hipStream_t stream)
{
    (void)in_sizes; (void)n_in; (void)out_size; (void)ws_size;

    const float* x      = (const float*)d_in[0];
    const float* w_a    = (const float*)d_in[1];
    const float* g_a    = (const float*)d_in[2];
    const float* b_a    = (const float*)d_in[3];
    const float* alpha1 = (const float*)d_in[4];
    const float* w_b    = (const float*)d_in[5];
    const float* g_b    = (const float*)d_in[6];
    const float* b_b    = (const float*)d_in[7];
    const float* alpha2 = (const float*)d_in[8];

    char* ws = (char*)d_ws;
    _Float16* fragA = (_Float16*)(ws + OFF_WFRAG_A);
    _Float16* fragB = (_Float16*)(ws + OFF_WFRAG_B);
    float* stats  = (float*)(ws + OFF_STATS);
    float* sum1 = stats,       *sq1 = stats + 64;
    float* sum2 = stats + 128, *sq2 = stats + 192;
    float* sb     = (float*)(ws + OFF_SCALEB);
    float* scale1 = sb,        *bias1 = sb + 64;
    float* scale2 = sb + 128,  *bias2 = sb + 192;
    _Float16* raw1 = (_Float16*)(ws + OFF_RAW1);
    _Float16* raw2 = (_Float16*)(ws + OFF_RAW2);
    float* out = (float*)d_out;

    const size_t total = (size_t)N_IMG * C * H * W;   // 33,554,432
    const dim3 convGrid(H, N_IMG);                    // (128, 32)

    zero_stats_kernel<<<1, 256, 0, stream>>>(stats);
    prep_weights_kernel<<<2, 256, 0, stream>>>(w_a, w_b, fragA, fragB);

    conv3x3_wmma_kernel<true><<<convGrid, 256, 0, stream>>>(
        (const void*)x, fragA, raw1, sum1, sq1);
    bn_finalize_kernel<<<1, 64, 0, stream>>>(sum1, sq1, g_a, b_a, scale1, bias1);
    bn_act_kernel<<<(unsigned)(total / (256 * 8)), 256, 0, stream>>>(
        raw1, scale1, bias1, alpha1);

    conv3x3_wmma_kernel<false><<<convGrid, 256, 0, stream>>>(
        (const void*)raw1, fragB, raw2, sum2, sq2);
    bn_finalize_kernel<<<1, 64, 0, stream>>>(sum2, sq2, g_b, b_b, scale2, bias2);
    residual_out_kernel<<<(unsigned)(total / (256 * 4)), 256, 0, stream>>>(
        x, raw2, scale2, bias2, alpha2, out);
}
